// GroupedQueryAttention_74002286510734
// MI455X (gfx1250) — matrix operand order repaired and checked by reading, not compiled, not run on hardware
//
#include <hip/hip_runtime.h>
#include <hip/hip_bf16.h>
#include <math.h>
#include <stdint.h>

typedef __attribute__((ext_vector_type(16))) _Float16 v16h;
typedef __attribute__((ext_vector_type(8)))  _Float16 v8h;
typedef __attribute__((ext_vector_type(8)))  float    v8f;

#define D_MODEL 1024
#define N_HEADS 16
#define N_KV    4
#define HD      64
#define BB      2
#define TT      2048
#define M_TOT   (BB*TT)          // 4096 rows
#define KV_DIM  (N_KV*HD)        // 256
#define SCALE   0.125f           // 1/sqrt(64)

// B-tile staged in LDS: 64 n-rows x 32 k (f16), row padded to 40 halves
// (80 B) so column reads are bank-conflict free.  Two buffers (10 KB).
#define BROW    40
#define BBUF_BYTES (64u * BROW * 2u)

// ---------------------------------------------------------------- helpers

static __device__ __forceinline__ v8f wmma_f16(v16h a, v16h b, v8f c) {
  return __builtin_amdgcn_wmma_f32_16x16x32_f16(
      /*neg_a=*/false, a, /*neg_b=*/false, b,
      /*c_mod=*/(short)0, c, /*reuse_a=*/false, /*reuse_b=*/false);
}

// Per-lane 16B global -> LDS async copy (ASYNCcnt-tracked DMA path).
static __device__ __forceinline__ void async_b128(unsigned lds_off,
                                                  const void* gaddr) {
  asm volatile("global_load_async_to_lds_b128 %0, %1, off"
               :: "v"(lds_off), "v"((unsigned long long)(uintptr_t)gaddr)
               : "memory");
}

// A-fragment (16x32, MxK) per-lane load from a row-major matrix.
// lane = (half<<4)|m ; VGPR i holds K = half*16 + 2*i {,+1}.
static __device__ __forceinline__ v16h load_a_frag(const _Float16* rowBase,
                                                   int k0, int hlf) {
  v8h lo = *reinterpret_cast<const v8h*>(rowBase + k0 + hlf * 16);
  v8h hi = *reinterpret_cast<const v8h*>(rowBase + k0 + hlf * 16 + 8);
  v16h r;
#pragma unroll
  for (int i = 0; i < 8; ++i) { r[i] = lo[i]; r[i + 8] = hi[i]; }
  return r;
}

static __device__ __forceinline__ float rowmax16(float v) {
  v = fmaxf(v, __shfl_xor(v, 1));
  v = fmaxf(v, __shfl_xor(v, 2));
  v = fmaxf(v, __shfl_xor(v, 4));
  v = fmaxf(v, __shfl_xor(v, 8));
  return v;
}
static __device__ __forceinline__ float rowsum16(float v) {
  v += __shfl_xor(v, 1);
  v += __shfl_xor(v, 2);
  v += __shfl_xor(v, 4);
  v += __shfl_xor(v, 8);
  return v;
}

// ---------------------------------------------------------------- converts

__global__ void cvt_f32_f16(const float* __restrict__ src,
                            _Float16* __restrict__ dst, int n) {
  int i = blockIdx.x * blockDim.x + threadIdx.x;
  if (i < n) dst[i] = (_Float16)src[i];
}

// src is (K x N) row-major fp32; dst becomes (N x K) row-major f16.
__global__ void transpose_cvt(const float* __restrict__ src,
                              _Float16* __restrict__ dst, int K, int N) {
  int i = blockIdx.x * blockDim.x + threadIdx.x;
  if (i < N * K) {
    int n = i / K, k = i - n * K;
    dst[(size_t)n * K + k] = (_Float16)src[(size_t)k * N + n];
  }
}

// ---------------------------------------------------------------- QKV GEMM
// A = xh (4096 x 1024) f16, Bt = wqkvT (1536 x 1024) f16 (pre-transposed).
// Block = 4 waves = 64 rows x 64 cols.  B-tile shared through LDS via
// double-buffered async copies; each wave computes a 16x64 tile.

__global__ __launch_bounds__(128) void qkv_gemm(
    const _Float16* __restrict__ xh, const _Float16* __restrict__ wt,
    const float* __restrict__ bq, const float* __restrict__ bk,
    const float* __restrict__ bv, _Float16* __restrict__ qh,
    _Float16* __restrict__ kh, _Float16* __restrict__ vtr) {
  __shared__ alignas(32) _Float16 btile[2][64 * BROW];

  int tid = threadIdx.x, wid = tid >> 5, lane = tid & 31;
  int hlf = lane >> 4, sub = lane & 15;
  int blk = blockIdx.x;                    // 0..1535
  int mb = blk / 24, nt = blk - mb * 24;
  int row0 = mb * 64 + wid * 16, n0 = nt * 64;

  const _Float16* arow = xh + (size_t)(row0 + sub) * D_MODEL;

  // this thread's 2 copy pieces: 256 x 16B pieces per k-step, 128 threads
  int i0 = tid, i1 = tid + 128;
  int nl0 = i0 >> 2, pc0 = i0 & 3;
  int nl1 = i1 >> 2, pc1 = i1 & 3;
  const _Float16* gsrc0 = wt + (size_t)(n0 + nl0) * D_MODEL + pc0 * 8;
  const _Float16* gsrc1 = wt + (size_t)(n0 + nl1) * D_MODEL + pc1 * 8;
  unsigned ldst0 = (unsigned)(uintptr_t)&btile[0][nl0 * BROW + pc0 * 8];
  unsigned ldst1 = (unsigned)(uintptr_t)&btile[0][nl1 * BROW + pc1 * 8];

  async_b128(ldst0, gsrc0);                // prologue: k0 = 0 -> buffer 0
  async_b128(ldst1, gsrc1);

  v8f acc[4] = {};
  for (int k0 = 0; k0 < D_MODEL; k0 += 32) {
    unsigned buf = (unsigned)(k0 >> 5) & 1u;
    if (k0 + 32 < D_MODEL) {               // prefetch next tile (other buffer)
      unsigned nb = (buf ^ 1u) * BBUF_BYTES;
      async_b128(ldst0 + nb, gsrc0 + k0 + 32);
      async_b128(ldst1 + nb, gsrc1 + k0 + 32);
      asm volatile("s_wait_asynccnt 0x2" ::: "memory");
    } else {
      asm volatile("s_wait_asynccnt 0x0" ::: "memory");
    }
    __syncthreads();                       // tile visible to all 4 waves

    v16h a = load_a_frag(arow, k0, hlf);
    const _Float16* lb = &btile[buf][0];
#pragma unroll
    for (int j = 0; j < 4; ++j) {
      v16h bf = *reinterpret_cast<const v16h*>(lb + (j * 16 + sub) * BROW + hlf * 16);
      acc[j] = wmma_f16(a, bf, acc[j]);
    }
    __syncthreads();                       // protect buffer before overwrite
  }

#pragma unroll
  for (int j = 0; j < 4; ++j) {
#pragma unroll
    for (int e = 0; e < 8; ++e) {
      int nc = n0 + j * 16 + sub;
      int gm = row0 + e + (hlf << 3);
      int b = gm >> 11, t = gm & (TT - 1);
      float bias = nc < D_MODEL ? bq[nc]
                  : (nc < D_MODEL + KV_DIM ? bk[nc - D_MODEL]
                                           : bv[nc - D_MODEL - KV_DIM]);
      float val = acc[j][e] + bias;
      if (nc < D_MODEL) {
        int h = nc >> 6, d = nc & 63;
        qh[(((size_t)b * N_HEADS + h) * TT + t) * HD + d] = (_Float16)val;
      } else if (nc < D_MODEL + KV_DIM) {
        int c = nc - D_MODEL, kv = c >> 6, d = c & 63;
        kh[(((size_t)b * N_KV + kv) * TT + t) * HD + d] = (_Float16)val;
      } else {
        int c = nc - D_MODEL - KV_DIM, kv = c >> 6, d = c & 63;
        vtr[(((size_t)b * N_KV + kv) * HD + d) * TT + t] = (_Float16)val;
      }
    }
  }
}

// ---------------------------------------------------------------- attention
// One wave per (b, h, 16-row Q tile). Streams 32-col KV blocks, online
// softmax in fp32, P converted C->A layout through per-wave LDS.

__global__ __launch_bounds__(128) void attn_kernel(
    const _Float16* __restrict__ qh, const _Float16* __restrict__ kh,
    const _Float16* __restrict__ vtr, _Float16* __restrict__ aout) {
  __shared__ alignas(32) _Float16 lds_p[4][16 * 32];

  int tid = threadIdx.x, wid = tid >> 5, lane = tid & 31;
  int hlf = lane >> 4, sub = lane & 15;
  int gw = blockIdx.x * 4 + wid;            // 0..4095
  int b = gw >> 11;
  int rem = gw & 2047;
  int h = rem >> 7;
  int q_base = (rem & 127) << 4;
  int kvh = h >> 2;                          // NUM_REP = 4
  int q_lim = q_base + 15;

  const _Float16* qrow = qh + (((size_t)b * N_HEADS + h) * TT + q_base + sub) * HD;
  v16h aq0 = load_a_frag(qrow, 0, hlf);
  v16h aq1 = load_a_frag(qrow, 32, hlf);

  const _Float16* kbase = kh + ((size_t)b * N_KV + kvh) * TT * HD;
  const _Float16* vbase = vtr + ((size_t)b * N_KV + kvh) * HD * TT;

  float m_r[8], l_r[8];
  v8f acc[4] = {};
#pragma unroll
  for (int e = 0; e < 8; ++e) { m_r[e] = -1e30f; l_r[e] = 0.0f; }

  for (int s = 0; s <= q_lim; s += 32) {
    bool do_hi = (s + 16 <= q_lim);          // wave-uniform

    // S0 = Q @ K^T  (cols s..s+15)
    const _Float16* kr0 = kbase + (size_t)(s + sub) * HD;
    v8f c0 = {};
    c0 = wmma_f16(aq0, *reinterpret_cast<const v16h*>(kr0 + hlf * 16), c0);
    c0 = wmma_f16(aq1, *reinterpret_cast<const v16h*>(kr0 + 32 + hlf * 16), c0);
    v8f c1 = {};
    if (do_hi) {                             // cols s+16..s+31
      const _Float16* kr1 = kbase + (size_t)(s + 16 + sub) * HD;
      c1 = wmma_f16(aq0, *reinterpret_cast<const v16h*>(kr1 + hlf * 16), c1);
      c1 = wmma_f16(aq1, *reinterpret_cast<const v16h*>(kr1 + 32 + hlf * 16), c1);
    }
    if (s + 32 <= q_lim)
      __builtin_prefetch(kbase + (size_t)(s + 32 + sub) * HD, 0, 1);

    // online softmax; write P (f16) to LDS in C layout
#pragma unroll
    for (int e = 0; e < 8; ++e) {
      int row = q_base + e + (hlf << 3);
      float s0 = c0[e] * SCALE;
      float s1 = c1[e] * SCALE;
      if (s + sub > row)      s0 = -1e30f;   // causal mask
      if (s + 16 + sub > row) s1 = -1e30f;
      float bmax = rowmax16(fmaxf(s0, s1));
      float m_new = fmaxf(m_r[e], bmax);
      float f  = __expf(m_r[e] - m_new);
      float p0 = __expf(s0 - m_new);
      float p1 = __expf(s1 - m_new);
      float rs = rowsum16(p0 + p1);
      l_r[e] = l_r[e] * f + rs;
      m_r[e] = m_new;
#pragma unroll
      for (int j = 0; j < 4; ++j) acc[j][e] *= f;
      int lrow = e + (hlf << 3);
      lds_p[wid][lrow * 32 + sub]      = (_Float16)p0;
      lds_p[wid][lrow * 32 + 16 + sub] = (_Float16)p1;
    }
    asm volatile("s_wait_dscnt 0x0" ::: "memory");

    // reload P as an A-fragment (row = sub, chunks at half*16 / half*16+8)
    v16h pf;
    {
      v8h lo = *reinterpret_cast<const v8h*>(&lds_p[wid][sub * 32 + hlf * 16]);
      v8h hi = *reinterpret_cast<const v8h*>(&lds_p[wid][sub * 32 + hlf * 16 + 8]);
#pragma unroll
      for (int i = 0; i < 8; ++i) { pf[i] = lo[i]; pf[i + 8] = hi[i]; }
    }
    asm volatile("s_wait_dscnt 0x0" ::: "memory");

    // O += P @ V   (V stored transposed: contiguous along T)
    int tb = s + (do_hi ? hlf * 16 : 0);     // clamp when upper half is all-zero P
#pragma unroll
    for (int j = 0; j < 4; ++j) {
      const _Float16* vr = vbase + (size_t)(j * 16 + sub) * TT + tb;
      acc[j] = wmma_f16(pf, *reinterpret_cast<const v16h*>(vr), acc[j]);
    }
  }

  // epilogue: normalize and store (b, t, h*64+d) f16
#pragma unroll
  for (int e = 0; e < 8; ++e) {
    float inv = 1.0f / l_r[e];
    int t = q_base + e + (hlf << 3);
    _Float16* orow = aout + ((size_t)b * TT + t) * D_MODEL + h * HD;
#pragma unroll
    for (int j = 0; j < 4; ++j)
      orow[j * 16 + sub] = (_Float16)(acc[j][e] * inv);
  }
}

// ---------------------------------------------------------------- out proj
// out = aout(4096x1024 f16) @ Wo + bo, fp32 result straight to d_out.
// Same LDS-staged, async double-buffered B-tile scheme as qkv_gemm.

__global__ __launch_bounds__(128) void out_gemm(
    const _Float16* __restrict__ ah, const _Float16* __restrict__ wot,
    const float* __restrict__ bo, float* __restrict__ out) {
  __shared__ alignas(32) _Float16 btile[2][64 * BROW];

  int tid = threadIdx.x, wid = tid >> 5, lane = tid & 31;
  int hlf = lane >> 4, sub = lane & 15;
  int blk = blockIdx.x;                     // 0..1023
  int mb = blk >> 4, nt = blk & 15;
  int row0 = mb * 64 + wid * 16, n0 = nt * 64;

  const _Float16* arow = ah + (size_t)(row0 + sub) * D_MODEL;

  int i0 = tid, i1 = tid + 128;
  int nl0 = i0 >> 2, pc0 = i0 & 3;
  int nl1 = i1 >> 2, pc1 = i1 & 3;
  const _Float16* gsrc0 = wot + (size_t)(n0 + nl0) * D_MODEL + pc0 * 8;
  const _Float16* gsrc1 = wot + (size_t)(n0 + nl1) * D_MODEL + pc1 * 8;
  unsigned ldst0 = (unsigned)(uintptr_t)&btile[0][nl0 * BROW + pc0 * 8];
  unsigned ldst1 = (unsigned)(uintptr_t)&btile[0][nl1 * BROW + pc1 * 8];

  async_b128(ldst0, gsrc0);
  async_b128(ldst1, gsrc1);

  v8f acc[4] = {};
  for (int k0 = 0; k0 < D_MODEL; k0 += 32) {
    unsigned buf = (unsigned)(k0 >> 5) & 1u;
    if (k0 + 32 < D_MODEL) {
      unsigned nb = (buf ^ 1u) * BBUF_BYTES;
      async_b128(ldst0 + nb, gsrc0 + k0 + 32);
      async_b128(ldst1 + nb, gsrc1 + k0 + 32);
      asm volatile("s_wait_asynccnt 0x2" ::: "memory");
    } else {
      asm volatile("s_wait_asynccnt 0x0" ::: "memory");
    }
    __syncthreads();

    v16h a = load_a_frag(arow, k0, hlf);
    const _Float16* lb = &btile[buf][0];
#pragma unroll
    for (int j = 0; j < 4; ++j) {
      v16h bf = *reinterpret_cast<const v16h*>(lb + (j * 16 + sub) * BROW + hlf * 16);
      acc[j] = wmma_f16(a, bf, acc[j]);
    }
    __syncthreads();
  }

#pragma unroll
  for (int j = 0; j < 4; ++j) {
#pragma unroll
    for (int e = 0; e < 8; ++e) {
      int nc = n0 + j * 16 + sub;
      int gm = row0 + e + (hlf << 3);
      out[(size_t)gm * D_MODEL + nc] = acc[j][e] + bo[nc];
    }
  }
}

// ---------------------------------------------------------------- launcher

extern "C" void kernel_launch(void* const* d_in, const int* in_sizes, int n_in,
                              void* d_out, int out_size, void* d_ws, size_t ws_size,
                              hipStream_t stream) {
  const float* x  = (const float*)d_in[0];
  // d_in[1] = mask (causal, implemented analytically)
  const float* Wq = (const float*)d_in[2];
  const float* bq = (const float*)d_in[3];
  const float* Wk = (const float*)d_in[4];
  const float* bk = (const float*)d_in[5];
  const float* Wv = (const float*)d_in[6];
  const float* bv = (const float*)d_in[7];
  const float* Wo = (const float*)d_in[8];
  const float* bo = (const float*)d_in[9];
  float* out = (float*)d_out;

  char* ws = (char*)d_ws;
  _Float16* xh     = (_Float16*)(ws);                         //  8 MB
  _Float16* wqkvT  = (_Float16*)(ws + 8388608);               //  3 MB (1536x1024)
  _Float16* wot    = (_Float16*)(ws + 11534336);              //  2 MB (1024x1024)
  _Float16* qh     = (_Float16*)(ws + 13631488);              //  8 MB (b,h,t,d)
  _Float16* kh     = (_Float16*)(ws + 22020096);              //  2 MB (b,kvh,t,d)
  _Float16* vtr    = (_Float16*)(ws + 24117248);              //  2 MB (b,kvh,d,t)
  _Float16* aout   = (_Float16*)(ws + 26214400);              //  8 MB (b,t,h*64+d)

  // fp32 -> f16 conversions / weight transposes
  cvt_f32_f16<<<(M_TOT * D_MODEL + 255) / 256, 256, 0, stream>>>(x, xh, M_TOT * D_MODEL);
  transpose_cvt<<<(D_MODEL * D_MODEL + 255) / 256, 256, 0, stream>>>(Wq, wqkvT, D_MODEL, D_MODEL);
  transpose_cvt<<<(KV_DIM * D_MODEL + 255) / 256, 256, 0, stream>>>(Wk, wqkvT + (size_t)D_MODEL * D_MODEL, D_MODEL, KV_DIM);
  transpose_cvt<<<(KV_DIM * D_MODEL + 255) / 256, 256, 0, stream>>>(Wv, wqkvT + (size_t)(D_MODEL + KV_DIM) * D_MODEL, D_MODEL, KV_DIM);
  transpose_cvt<<<(D_MODEL * D_MODEL + 255) / 256, 256, 0, stream>>>(Wo, wot, D_MODEL, D_MODEL);

  // fused QKV projection: 1536 blocks x (64 rows x 64 cols)
  qkv_gemm<<<1536, 128, 0, stream>>>(xh, wqkvT, bq, bk, bv, qh, kh, vtr);

  // flash attention: 4096 waves (b x h x 128 q-tiles), 4 waves/block
  attn_kernel<<<1024, 128, 0, stream>>>(qh, kh, vtr, aout);

  // output projection: 1024 blocks x (64 rows x 64 cols)
  out_gemm<<<1024, 128, 0, stream>>>(aout, wot, bo, out);
}
